// ACNet_14388140442037
// MI455X (gfx1250) — matrix operand order repaired and checked by reading, not compiled, not run on hardware
//
#include <hip/hip_runtime.h>
#include <hip/hip_bf16.h>

typedef __attribute__((ext_vector_type(16))) _Float16 v16h;
typedef __attribute__((ext_vector_type(8)))  _Float16 v8h;
typedef __attribute__((ext_vector_type(8)))  float    v8f;
typedef __attribute__((ext_vector_type(4)))  float    v4f;

#define N_NODES 50000
#define N_EDGES 800000
#define N_GRAPH 64
#define HID     128

// ---------------------------------------------------------------- helpers

__device__ __forceinline__ int laneid() { return (int)(threadIdx.x & 31); }

__device__ __forceinline__ v8f wmma16(v16h a, v16h b, v8f c) {
  // D = A(16x32 f16) * B(32x16 f16) + C(16x16 f32)
  return __builtin_amdgcn_wmma_f32_16x16x32_f16(false, a, false, b, (short)0, c,
                                                false, false);
}

// A-matrix layout (16-bit, 16x32): lanes 0-15 hold row M=lane, K in {0..7,16..23};
// lanes 16-31 hold row M=lane-16, K in {8..15,24..31}.
__device__ __forceinline__ v16h a_from_f16(const _Float16* row, int k0) {
  int kb = k0 + ((laneid() < 16) ? 0 : 8);
  v8h lo = *(const v8h*)(row + kb);
  v8h hi = *(const v8h*)(row + kb + 16);
  v16h a;
#pragma unroll
  for (int i = 0; i < 8; ++i) { a[i] = lo[i]; a[i + 8] = hi[i]; }
  return a;
}

__device__ __forceinline__ v16h a_from_f32(const float* row, int k0, float scale) {
  int kb = k0 + ((laneid() < 16) ? 0 : 8);
  v4f l0 = *(const v4f*)(row + kb);
  v4f l1 = *(const v4f*)(row + kb + 4);
  v4f h0 = *(const v4f*)(row + kb + 16);
  v4f h1 = *(const v4f*)(row + kb + 20);
  v16h a;
#pragma unroll
  for (int i = 0; i < 4; ++i) {
    a[i]      = (_Float16)(l0[i] * scale);
    a[i + 4]  = (_Float16)(l1[i] * scale);
    a[i + 8]  = (_Float16)(h0[i] * scale);
    a[i + 12] = (_Float16)(h1[i] * scale);
  }
  return a;
}

// Pre-swizzled B blocks: block (nt,kc) holds 32 lanes x 16 halves contiguously.
__device__ __forceinline__ v16h b_block(const _Float16* P, int nt, int kc, int KC) {
  return *(const v16h*)(P + (((size_t)(nt * KC + kc) * 32 + laneid()) << 4));
}

// C/D layout (f32 16x16): VGPR r, lane l -> row r + (l<16?0:8), col = l&15 (+nt*16)
__device__ __forceinline__ v8f bias_c(const float* bias, int nt) {
  float bv = bias[nt * 16 + (laneid() & 15)];
  v8f c;
#pragma unroll
  for (int r = 0; r < 8; ++r) c[r] = bv;
  return c;
}

// order-preserving float<->uint encoding for atomic max
__device__ __forceinline__ unsigned enc_f32(float f) {
  unsigned u = __float_as_uint(f);
  return (u & 0x80000000u) ? ~u : (u | 0x80000000u);
}
__device__ __forceinline__ float dec_f32(unsigned u) {
  unsigned b = (u & 0x80000000u) ? (u ^ 0x80000000u) : ~u;
  return __uint_as_float(b);
}

// ---------------------------------------------------------------- prep kernels

__global__ void cvt_f16_kernel(const float* __restrict__ src,
                               _Float16* __restrict__ dst, int n) {
  int i = blockIdx.x * 256 + threadIdx.x;
  if (i < n) dst[i] = (_Float16)src[i];
}

// W row-major [K, 128] f32 -> packed B-layout f16 blocks
__global__ void pack_weight_kernel(const float* __restrict__ W,
                                   _Float16* __restrict__ P, int K) {
  int idx = blockIdx.x * 256 + threadIdx.x;
  int KC = K / 32;
  int total = 8 * KC * 32 * 16;  // 8 n-tiles (Nfull=128)
  if (idx >= total) return;
  int j = idx & 15;
  int l = (idx >> 4) & 31;
  int blk = idx >> 9;
  int kc = blk % KC;
  int nt = blk / KC;
  int n = nt * 16 + (l & 15);
  int kk = kc * 32 + ((l < 16) ? 0 : 8) + j + ((j < 8) ? 0 : 8);  // B: lanes 0-15 K={0..7,16..23}, 16-31 K={8..15,24..31}
  P[idx] = (_Float16)W[(size_t)kk * 128 + n];
}

__global__ void zero_kernel(unsigned* __restrict__ p, size_t n) {
  size_t i = (size_t)blockIdx.x * 256 + threadIdx.x;
  if (i < n) p[i] = 0u;
}

__global__ void count_kernel(const int* __restrict__ batch,
                             float* __restrict__ gcnt, int n) {
  int i = blockIdx.x * 256 + threadIdx.x;
  if (i < n) atomicAdd(&gcnt[batch[i]], 1.0f);
}

// ---------------------------------------------------------------- feat / keyv

__global__ __launch_bounds__(128) void featkey_kernel(
    const _Float16* __restrict__ x16,
    const _Float16* __restrict__ WnP, const float* __restrict__ bn,
    const _Float16* __restrict__ WkP, const float* __restrict__ bk,
    float* __restrict__ feat, float* __restrict__ keyv) {
  int wave = threadIdx.x >> 5, lane = threadIdx.x & 31;
  int tile = blockIdx.x * 4 + wave;
  const int NT = N_NODES / 16;
  if (tile >= NT) return;
  const _Float16* P = blockIdx.y ? WkP : WnP;
  const float* bias = blockIdx.y ? bk : bn;
  float* out = blockIdx.y ? keyv : feat;
  int m = lane & 15, rbase = (lane < 16) ? 0 : 8;
  const _Float16* xrow = x16 + (size_t)(tile * 16 + m) * HID;

  v8f acc[8];
#pragma unroll
  for (int nt = 0; nt < 8; ++nt) acc[nt] = bias_c(bias, nt);
#pragma unroll
  for (int kc = 0; kc < 4; ++kc) {
    v16h a = a_from_f16(xrow, kc * 32);
#pragma unroll
    for (int nt = 0; nt < 8; ++nt)
      acc[nt] = wmma16(a, b_block(P, nt, kc, 4), acc[nt]);
  }
#pragma unroll
  for (int nt = 0; nt < 8; ++nt) {
    int col = nt * 16 + m;
#pragma unroll
    for (int r = 0; r < 8; ++r)
      out[(size_t)(tile * 16 + rbase + r) * HID + col] = acc[nt][r];
  }
}

// ---------------------------------------------------------------- edge model

__global__ __launch_bounds__(128) void edge_kernel(
    const _Float16* __restrict__ x16, const float* __restrict__ edge_attr,
    const int* __restrict__ S, const int* __restrict__ R,
    const _Float16* __restrict__ WeP, const float* __restrict__ be,
    const _Float16* __restrict__ WcP, const float* __restrict__ bc,
    const float* __restrict__ keyv,
    float* __restrict__ e_out, float* __restrict__ logits,
    unsigned* __restrict__ nodemax) {
  __shared__ _Float16 lds_e[4][16 * HID];
  int wave = threadIdx.x >> 5, lane = threadIdx.x & 31;
  int tile = blockIdx.x * 4 + wave;
  if (tile >= N_EDGES / 16) return;
  int m = lane & 15, rbase = (lane < 16) ? 0 : 8;

  int s = S[tile * 16 + m];
  const _Float16* xrow = x16 + (size_t)s * HID;
  const float* earow = edge_attr + (size_t)(tile * 16 + m) * 64;

  // ---- e = relu([x[S] | edge_attr] @ We + be), K = 192
  v8f acc[8];
#pragma unroll
  for (int nt = 0; nt < 8; ++nt) acc[nt] = bias_c(be, nt);
#pragma unroll
  for (int kc = 0; kc < 6; ++kc) {
    v16h a = (kc < 4) ? a_from_f16(xrow, kc * 32)
                      : a_from_f32(earow, (kc - 4) * 32, 1.0f);
#pragma unroll
    for (int nt = 0; nt < 8; ++nt)
      acc[nt] = wmma16(a, b_block(WeP, nt, kc, 6), acc[nt]);
  }
  // relu; write f32 e to output; stage f16 e in LDS (D->A cross-lane re-layout)
#pragma unroll
  for (int nt = 0; nt < 8; ++nt) {
    int col = nt * 16 + m;
#pragma unroll
    for (int r = 0; r < 8; ++r) {
      float v = acc[nt][r]; v = v > 0.0f ? v : 0.0f;
      e_out[(size_t)(tile * 16 + rbase + r) * HID + col] = v;
      lds_e[wave][(rbase + r) * HID + col] = (_Float16)v;
    }
  }
  asm volatile("s_wait_dscnt 0" ::: "memory");

  // ---- code = e @ Wc + bc, K = 128
  v8f cc[8];
#pragma unroll
  for (int nt = 0; nt < 8; ++nt) cc[nt] = bias_c(bc, nt);
  const _Float16* erow16 = &lds_e[wave][m * HID];
#pragma unroll
  for (int kc = 0; kc < 4; ++kc) {
    v16h a = a_from_f16(erow16, kc * 32);
#pragma unroll
    for (int nt = 0; nt < 8; ++nt)
      cc[nt] = wmma16(a, b_block(WcP, nt, kc, 4), cc[nt]);
  }

  // ---- logits[e] = dot(keyv[R[e]], code[e])
  int Rv[8];
#pragma unroll
  for (int r = 0; r < 8; ++r) Rv[r] = R[tile * 16 + rbase + r];
  float part[8];
#pragma unroll
  for (int r = 0; r < 8; ++r) part[r] = 0.0f;
#pragma unroll
  for (int nt = 0; nt < 8; ++nt) {
    int col = nt * 16 + m;
#pragma unroll
    for (int r = 0; r < 8; ++r)
      part[r] += cc[nt][r] * keyv[(size_t)Rv[r] * HID + col];
  }
#pragma unroll
  for (int mask = 8; mask >= 1; mask >>= 1) {
#pragma unroll
    for (int r = 0; r < 8; ++r) part[r] += __shfl_xor(part[r], mask, 16);
  }
  if (m == 0) {
#pragma unroll
    for (int r = 0; r < 8; ++r) {
      logits[tile * 16 + rbase + r] = part[r];
      atomicMax(nodemax + Rv[r], enc_f32(part[r]));
    }
  }
}

// ---------------------------------------------------------------- softmax/agg

__global__ void softmax_ex_kernel(float* __restrict__ logits,
                                  const int* __restrict__ R,
                                  const unsigned* __restrict__ nodemax,
                                  float* __restrict__ den) {
  int i = blockIdx.x * 256 + threadIdx.x;
  if (i >= N_EDGES) return;
  int r = R[i];
  float ex = expf(logits[i] - dec_f32(nodemax[r]));
  logits[i] = ex;  // reuse buffer as ex
  atomicAdd(&den[r], ex);
}

__global__ void aggregate_kernel(const float* __restrict__ e_out,
                                 const float* __restrict__ ex,
                                 const float* __restrict__ den,
                                 const int* __restrict__ R,
                                 float* __restrict__ agg) {
  int i = blockIdx.x * 256 + threadIdx.x;  // over E*HID (< 2^31)
  if (i >= N_EDGES * HID) return;
  int e = i >> 7, c = i & 127;
  int r = R[e];
  float w = ex[e] / den[r];
  atomicAdd(&agg[(size_t)r * HID + c], e_out[i] * w);
}

// ---------------------------------------------------------------- node model

__global__ __launch_bounds__(128) void node_kernel(
    const float* __restrict__ feat, const float* __restrict__ agg,
    const float* __restrict__ glob, const int* __restrict__ batch,
    const _Float16* __restrict__ Wn2P, const float* __restrict__ bn2,
    float* __restrict__ nodes_out, float* __restrict__ gsum) {
  int wave = threadIdx.x >> 5, lane = threadIdx.x & 31;
  int tile = blockIdx.x * 4 + wave;
  const int NT = N_NODES / 16;
  if (tile >= NT) return;
  int m = lane & 15, rbase = (lane < 16) ? 0 : 8;
  int node = tile * 16 + m;
  const float* frow = feat + (size_t)node * HID;
  const float* arow = agg + (size_t)node * HID;
  const float* grow = glob + (size_t)batch[node] * 64;
  int bv[8];
#pragma unroll
  for (int r = 0; r < 8; ++r) bv[r] = batch[tile * 16 + rbase + r];

  v8f acc[8];
#pragma unroll
  for (int nt = 0; nt < 8; ++nt) acc[nt] = bias_c(bn2, nt);
#pragma unroll
  for (int kc = 0; kc < 10; ++kc) {  // K = 320
    v16h a = (kc < 4)   ? a_from_f32(frow, kc * 32, 1.0f)
             : (kc < 8) ? a_from_f32(arow, (kc - 4) * 32, 1.0f)
                        : a_from_f32(grow, (kc - 8) * 32, 1.0f);
#pragma unroll
    for (int nt = 0; nt < 8; ++nt)
      acc[nt] = wmma16(a, b_block(Wn2P, nt, kc, 10), acc[nt]);
  }
#pragma unroll
  for (int nt = 0; nt < 8; ++nt) {
    int col = nt * 16 + m;
#pragma unroll
    for (int r = 0; r < 8; ++r) {
      float v = acc[nt][r]; v = v > 0.0f ? v : 0.0f;
      nodes_out[(size_t)(tile * 16 + rbase + r) * HID + col] = v;
      atomicAdd(&gsum[(size_t)bv[r] * HID + col], v);
    }
  }
}

// ---------------------------------------------------------------- global model

__global__ __launch_bounds__(128) void global_kernel(
    const float* __restrict__ glob, const float* __restrict__ gsum,
    const float* __restrict__ gcnt,
    const _Float16* __restrict__ WgP, const float* __restrict__ bg,
    float* __restrict__ globs_out) {
  int wave = threadIdx.x >> 5, lane = threadIdx.x & 31;
  int tile = wave;  // G/16 == 4 tiles, one block of 4 waves
  int m = lane & 15, rbase = (lane < 16) ? 0 : 8;
  int g = tile * 16 + m;
  float invc = 1.0f / fmaxf(gcnt[g], 1.0f);
  const float* grow = glob + (size_t)g * 64;
  const float* srow = gsum + (size_t)g * HID;

  v8f acc[8];
#pragma unroll
  for (int nt = 0; nt < 8; ++nt) acc[nt] = bias_c(bg, nt);
#pragma unroll
  for (int kc = 0; kc < 6; ++kc) {  // K = 192: glob(64) | mean(128)
    v16h a = (kc < 2) ? a_from_f32(grow, kc * 32, 1.0f)
                      : a_from_f32(srow, (kc - 2) * 32, invc);
#pragma unroll
    for (int nt = 0; nt < 8; ++nt)
      acc[nt] = wmma16(a, b_block(WgP, nt, kc, 6), acc[nt]);
  }
#pragma unroll
  for (int nt = 0; nt < 8; ++nt) {
    int col = nt * 16 + m;
#pragma unroll
    for (int r = 0; r < 8; ++r) {
      float v = acc[nt][r]; v = v > 0.0f ? v : 0.0f;
      globs_out[(size_t)(tile * 16 + rbase + r) * HID + col] = v;
    }
  }
}

// ---------------------------------------------------------------- launch

extern "C" void kernel_launch(void* const* d_in, const int* in_sizes, int n_in,
                              void* d_out, int out_size, void* d_ws, size_t ws_size,
                              hipStream_t stream) {
  const float* x         = (const float*)d_in[0];
  const float* edge_attr = (const float*)d_in[1];
  const float* glob      = (const float*)d_in[2];
  const int*   edge_idx  = (const int*)d_in[3];
  const int*   batch     = (const int*)d_in[4];
  const float* We  = (const float*)d_in[5];  const float* be  = (const float*)d_in[6];
  const float* Wc  = (const float*)d_in[7];  const float* bc  = (const float*)d_in[8];
  const float* Wk  = (const float*)d_in[9];  const float* bk  = (const float*)d_in[10];
  const float* Wn  = (const float*)d_in[11]; const float* bn  = (const float*)d_in[12];
  const float* Wn2 = (const float*)d_in[13]; const float* bn2 = (const float*)d_in[14];
  const float* Wg  = (const float*)d_in[15]; const float* bg  = (const float*)d_in[16];
  const int* S = edge_idx;
  const int* R = edge_idx + N_EDGES;

  float* nodes_out = (float*)d_out;
  float* e_out     = nodes_out + (size_t)N_NODES * HID;
  float* globs_out = e_out + (size_t)N_EDGES * HID;

  // workspace carving (256B aligned)
  char* w = (char*)d_ws;
  size_t off = 0;
  auto carve = [&](size_t bytes) -> char* {
    char* p = w + off;
    off += (bytes + 255) & ~(size_t)255;
    return p;
  };
  _Float16* x16  = (_Float16*)carve((size_t)N_NODES * HID * 2);
  _Float16* WeP  = (_Float16*)carve(192 * 128 * 2);
  _Float16* WcP  = (_Float16*)carve(128 * 128 * 2);
  _Float16* WkP  = (_Float16*)carve(128 * 128 * 2);
  _Float16* WnP  = (_Float16*)carve(128 * 128 * 2);
  _Float16* Wn2P = (_Float16*)carve(320 * 128 * 2);
  _Float16* WgP  = (_Float16*)carve(192 * 128 * 2);
  char* zbase = w + off;  // contiguous zero-init region
  unsigned* nodemax = (unsigned*)carve((size_t)N_NODES * 4);
  float* den  = (float*)carve((size_t)N_NODES * 4);
  float* agg  = (float*)carve((size_t)N_NODES * HID * 4);
  float* gsum = (float*)carve((size_t)N_GRAPH * HID * 4);
  float* gcnt = (float*)carve((size_t)N_GRAPH * 4);
  size_t zwords = ((w + off) - zbase) / 4;
  float* feat   = (float*)carve((size_t)N_NODES * HID * 4);
  float* keyv   = (float*)carve((size_t)N_NODES * HID * 4);
  float* logits = (float*)carve((size_t)N_EDGES * 4);
  (void)ws_size; (void)in_sizes; (void)n_in; (void)out_size;

  // 1) prep
  cvt_f16_kernel<<<(N_NODES * HID + 255) / 256, 256, 0, stream>>>(x, x16, N_NODES * HID);
  pack_weight_kernel<<<(8 * 6 * 512 + 255) / 256, 256, 0, stream>>>(We, WeP, 192);
  pack_weight_kernel<<<(8 * 4 * 512 + 255) / 256, 256, 0, stream>>>(Wc, WcP, 128);
  pack_weight_kernel<<<(8 * 4 * 512 + 255) / 256, 256, 0, stream>>>(Wk, WkP, 128);
  pack_weight_kernel<<<(8 * 4 * 512 + 255) / 256, 256, 0, stream>>>(Wn, WnP, 128);
  pack_weight_kernel<<<(8 * 10 * 512 + 255) / 256, 256, 0, stream>>>(Wn2, Wn2P, 320);
  pack_weight_kernel<<<(8 * 6 * 512 + 255) / 256, 256, 0, stream>>>(Wg, WgP, 192);
  zero_kernel<<<(unsigned)((zwords + 255) / 256), 256, 0, stream>>>((unsigned*)zbase, zwords);
  count_kernel<<<(N_NODES + 255) / 256, 256, 0, stream>>>(batch, gcnt, N_NODES);

  // 2) feat / keyv
  {
    dim3 grid((N_NODES / 16 + 3) / 4, 2);
    featkey_kernel<<<grid, 128, 0, stream>>>(x16, WnP, bn, WkP, bk, feat, keyv);
  }
  // 3) edge model + code + logits + per-node max
  edge_kernel<<<(N_EDGES / 16) / 4, 128, 0, stream>>>(
      x16, edge_attr, S, R, WeP, be, WcP, bc, keyv, e_out, logits, nodemax);
  // 4) segment softmax + weighted aggregation
  softmax_ex_kernel<<<(N_EDGES + 255) / 256, 256, 0, stream>>>(logits, R, nodemax, den);
  aggregate_kernel<<<(N_EDGES * HID) / 256, 256, 0, stream>>>(e_out, logits, den, R, agg);
  // 5) node model + per-graph sums
  node_kernel<<<(N_NODES / 16 + 3) / 4, 128, 0, stream>>>(
      feat, agg, glob, batch, Wn2P, bn2, nodes_out, gsum);
  // 6) global model
  global_kernel<<<1, 128, 0, stream>>>(glob, gsum, gcnt, WgP, bg, globs_out);
}
